// Backbone_30631706755945
// MI455X (gfx1250) — matrix operand order repaired and checked by reading, not compiled, not run on hardware
//
#include <hip/hip_runtime.h>

// ---------------------------------------------------------------------------
// CDNA5 (gfx1250) Point Transformer backbone.
// All dense math is bf16 WMMA (v_wmma_f32_16x16x32_bf16), fp32 accumulate.
// Each wave computes a 16x64 tile: one A fragment feeds 4 WMMAs (4 acc).
// ---------------------------------------------------------------------------

typedef __attribute__((ext_vector_type(16))) __bf16 v16bf;
typedef __attribute__((ext_vector_type(8)))  __bf16 v8bf;
typedef __attribute__((ext_vector_type(8)))  float  v8f;

#define DM 512
#define KNN 16
#define INV_BN 0.9999950000374997f       // 1/sqrt(1 + 1e-5)
#define INV_SQRT_DM 0.04419417382415922f // 1/sqrt(512)

// ---- fragment loaders -----------------------------------------------------
// A fragment (contiguous K order, same as B): lane half selects K sub-blocks
//   elements 0..7  <- K = k0 + half*16 + e        (8 consecutive)
//   elements 8..15 <- K = k0 + half*16 + 8 + e-8  (8 consecutive)
__device__ inline v16bf afrag_f32(const float* __restrict__ xrow, int k0, int half)
{
    const float4* p0 = (const float4*)(xrow + k0 + half * 16);
    const float4* p1 = (const float4*)(xrow + k0 + half * 16 + 8);
    float4 a0 = p0[0], a1 = p0[1], a2 = p1[0], a3 = p1[1];
    v16bf a;
    a[0]=(__bf16)a0.x;  a[1]=(__bf16)a0.y;  a[2]=(__bf16)a0.z;  a[3]=(__bf16)a0.w;
    a[4]=(__bf16)a1.x;  a[5]=(__bf16)a1.y;  a[6]=(__bf16)a1.z;  a[7]=(__bf16)a1.w;
    a[8]=(__bf16)a2.x;  a[9]=(__bf16)a2.y;  a[10]=(__bf16)a2.z; a[11]=(__bf16)a2.w;
    a[12]=(__bf16)a3.x; a[13]=(__bf16)a3.y; a[14]=(__bf16)a3.z; a[15]=(__bf16)a3.w;
    return a;
}

// B fragment: B[k][n] = W[n][k]; lane owns one output column, 16 consecutive K
__device__ inline v16bf bfrag_f32(const float* __restrict__ wr, int k0, int half)
{
    const float4* p = (const float4*)(wr + k0 + half * 16);
    float4 w0 = p[0], w1 = p[1], w2 = p[2], w3 = p[3];
    v16bf b;
    b[0]=(__bf16)w0.x;  b[1]=(__bf16)w0.y;  b[2]=(__bf16)w0.z;  b[3]=(__bf16)w0.w;
    b[4]=(__bf16)w1.x;  b[5]=(__bf16)w1.y;  b[6]=(__bf16)w1.z;  b[7]=(__bf16)w1.w;
    b[8]=(__bf16)w2.x;  b[9]=(__bf16)w2.y;  b[10]=(__bf16)w2.z; b[11]=(__bf16)w2.w;
    b[12]=(__bf16)w3.x; b[13]=(__bf16)w3.y; b[14]=(__bf16)w3.z; b[15]=(__bf16)w3.w;
    return b;
}

// A fragment out of LDS (already bf16): two 16-byte ds loads
__device__ inline v16bf afrag_lds(const __bf16* __restrict__ ar, int k0, int half)
{
    const v8bf* p0 = (const v8bf*)(ar + k0 + half * 16);
    const v8bf* p1 = (const v8bf*)(ar + k0 + half * 16 + 8);
    v8bf lo = p0[0], hi = p1[0];
    return __builtin_shufflevector(lo, hi, 0,1,2,3,4,5,6,7,8,9,10,11,12,13,14,15);
}

// ---------------------------------------------------------------------------
// Generic GEMM: Y[R,Cout] = epi( X[R,Cin](stride lda) @ W[Cout,Cin]^T )
// epi: (+bias) -> (*scale*INV_BN + shift) -> relu -> (+resid)
// One 16x64 tile per wave (4 accumulators reuse one A fragment).
// WVEC template hoists the W-row-alignment choice out of the K loop.
// ---------------------------------------------------------------------------
template<bool WVEC>
__device__ __forceinline__ void gemm_kloop(const float* __restrict__ xrow,
                                           const float* const* wr, int kmain,
                                           int half, v8f* acc)
{
    for (int k0 = 0; k0 < kmain; k0 += 32) {
        v16bf a = afrag_f32(xrow, k0, half);
#pragma unroll
        for (int s = 0; s < 4; ++s) {
            __builtin_prefetch(wr[s] + k0 + 32, 0, 1);   // speculative: safe past row end
            v16bf b;
            if (WVEC) {
                b = bfrag_f32(wr[s], k0, half);
            } else {
                const float* __restrict__ ws_ = wr[s] + k0 + half * 16;
#pragma unroll
                for (int e = 0; e < 16; ++e) b[e] = (__bf16)ws_[e];
            }
            acc[s] = __builtin_amdgcn_wmma_f32_16x16x32_bf16(false, a, false, b,
                                                             (short)0, acc[s], false, false);
        }
    }
}

__global__ __launch_bounds__(256) void gemm_wmma(
    const float* __restrict__ X, const float* __restrict__ W,
    const float* __restrict__ bias, const float* __restrict__ scale,
    const float* __restrict__ shift, const float* __restrict__ resid,
    float* __restrict__ Y, int R, int Cin, int lda, int Cout, int relu)
{
    const int lane = threadIdx.x & 31;
    const int wave = threadIdx.x >> 5;
    const int rt = (R + 15) >> 4;
    const int ct = (Cout + 63) >> 6;                 // 64-wide column tiles
    const long tile = (long)blockIdx.x * 8 + wave;
    if (tile >= (long)rt * ct) return;               // wave-uniform: EXEC stays all-1
    const int m0 = (int)(tile / ct) * 16;
    const int n0 = (int)(tile % ct) * 64;
    const int half = lane >> 4, r = lane & 15;

    const int arow = m0 + r;
    // Clamp pointers instead of masking loads: clamped rows produce values only
    // for output rows/cols that are never stored.
    const float* __restrict__ xrow = X + (long)(arow < R ? arow : R - 1) * lda;
    const float* wr[4];
#pragma unroll
    for (int s = 0; s < 4; ++s) {
        int wrow = n0 + s * 16 + r;
        wr[s] = W + (long)(wrow < Cout ? wrow : Cout - 1) * Cin;
    }

    v8f acc[4];
#pragma unroll
    for (int s = 0; s < 4; ++s) acc[s] = (v8f){0.f,0.f,0.f,0.f,0.f,0.f,0.f,0.f};

    const int kmain = Cin & ~31;
    if ((Cin & 3) == 0) gemm_kloop<true >(xrow, wr, kmain, half, acc);
    else                gemm_kloop<false>(xrow, wr, kmain, half, acc);

    if (kmain < Cin) {                               // ragged-K tail, branch-free
        v16bf a;
#pragma unroll
        for (int e = 0; e < 16; ++e) {
            int ka = kmain + half * 16 + e;
            float av = xrow[ka < Cin ? ka : (Cin - 1)];       // unconditional load
            a[e] = (__bf16)((ka < Cin) ? av : 0.f);           // v_cndmask select
        }
#pragma unroll
        for (int s = 0; s < 4; ++s) {
            v16bf b;
#pragma unroll
            for (int e = 0; e < 16; ++e) {
                int kb = kmain + half * 16 + e;
                float bv = wr[s][kb < Cin ? kb : (Cin - 1)];
                b[e] = (__bf16)((kb < Cin) ? bv : 0.f);
            }
            acc[s] = __builtin_amdgcn_wmma_f32_16x16x32_bf16(false, a, false, b,
                                                             (short)0, acc[s], false, false);
        }
    }

#pragma unroll
    for (int s = 0; s < 4; ++s) {
        const int col = n0 + s * 16 + r;
        if (col < Cout) {
            const float bb = bias  ? bias[col]  : 0.f;
            const float sc = scale ? scale[col] * INV_BN : 1.f;
            const float sh = shift ? shift[col] : 0.f;
#pragma unroll
            for (int j = 0; j < 8; ++j) {
                int rr = m0 + half * 8 + j;
                if (rr < R) {
                    float y = acc[s][j] + bb;
                    if (scale) y = y * sc + sh;
                    if (relu)  y = fmaxf(y, 0.f);
                    if (resid) y += resid[(long)rr * Cout + col];
                    Y[(long)rr * Cout + col] = y;
                }
            }
        }
    }
}

// ---------------------------------------------------------------------------
// Fused attention: each of 8 waves owns 64 of the 512 output columns.
// A (16x512, bf16) resident in LDS; one A fragment feeds 4 WMMAs per chunk.
// ---------------------------------------------------------------------------
__device__ inline void gemm16x64_lds(const __bf16* __restrict__ sA,
                                     const float* __restrict__ W,
                                     int n0, int lane, v8f* acc)
{
    const int half = lane >> 4, r = lane & 15;
    const __bf16* __restrict__ ar = sA + r * DM;
    const float* wr[4];
#pragma unroll
    for (int s = 0; s < 4; ++s) wr[s] = W + (long)(n0 + s * 16 + r) * DM;
    for (int k0 = 0; k0 < DM; k0 += 32) {
        v16bf a = afrag_lds(ar, k0, half);     // 2x b128 ds loads, reused 4x
#pragma unroll
        for (int s = 0; s < 4; ++s) {
            v16bf b = bfrag_f32(wr[s], k0, half);
            acc[s] = __builtin_amdgcn_wmma_f32_16x16x32_bf16(false, a, false, b,
                                                             (short)0, acc[s], false, false);
        }
    }
}

__device__ inline void store16_lds(__bf16* __restrict__ dst, const v8f& acc,
                                   const float* __restrict__ bias,
                                   int n0, int lane, bool relu)
{
    const int half = lane >> 4, r = lane & 15, col = n0 + r;
    const float bb = bias ? bias[col] : 0.f;
#pragma unroll
    for (int j = 0; j < 8; ++j) {
        float y = acc[j] + bb;
        if (relu) y = fmaxf(y, 0.f);
        dst[(half * 8 + j) * DM + col] = (__bf16)y;
    }
}

// One workgroup (256 thr, 8 waves) per query point:
//   h1 = relu(d1(cen - nbr));  pos = d2(h1)
//   t  = q - k_gather + pos;   a1 = relu(g1(t));  logits = g2(a1)
//   attn = softmax_K(logits/sqrt(512)); res = sum_K attn*(v_gather+pos)
// Keeps all (B,N,K,512) tensors out of HBM (~1 GB of traffic per block saved).
__global__ __launch_bounds__(256) void attn_fused(
    const float* __restrict__ xyz, const int* __restrict__ idx,
    const float* __restrict__ qb, const float* __restrict__ kb,
    const float* __restrict__ vb,
    const float* __restrict__ d1_w, const float* __restrict__ d1_b,
    const float* __restrict__ d2_w, const float* __restrict__ d2_b,
    const float* __restrict__ g1_w, const float* __restrict__ g1_b,
    const float* __restrict__ g2_w, const float* __restrict__ g2_b,
    float* __restrict__ res, int B, int N)
{
    __shared__ __bf16 sA[KNN * DM];   // h1, then a1
    __shared__ __bf16 sT[KNN * DM];   // t,  then attention logits
    __shared__ __bf16 sP[KNN * DM];   // pos
    __shared__ float  sQ[DM];
    __shared__ int    sIdx[KNN];
    __shared__ float  sCen[3];

    const int t  = threadIdx.x;
    const int pm = blockIdx.x;            // flat (b*N + m)
    const int b  = pm / N;
    const int lane = t & 31, wave = t >> 5;
    const int n0 = wave * 64;             // this wave's 64 columns

    if (t < KNN) sIdx[t] = idx[(long)pm * KNN + t];
    if (t < 3)   sCen[t] = xyz[(long)pm * 3 + t];
    for (int f = t; f < DM; f += 256) sQ[f] = qb[(long)pm * DM + f];
    __syncthreads();

    // ---- h1 = relu(d1 * (cen - nbr) + b) : K=3, plain VALU ----
    for (int e = t; e < KNN * DM; e += 256) {
        int k = e >> 9, c = e & (DM - 1);
        const float* np = xyz + ((long)b * N + sIdx[k]) * 3;
        float dx = sCen[0] - np[0], dy = sCen[1] - np[1], dz = sCen[2] - np[2];
        float h = d1_b[c] + dx * d1_w[c * 3] + dy * d1_w[c * 3 + 1] + dz * d1_w[c * 3 + 2];
        sA[e] = (__bf16)fmaxf(h, 0.f);
    }
    __syncthreads();

    // ---- pos = h1 @ d2^T + b ----
    {
        v8f acc[4];
#pragma unroll
        for (int s = 0; s < 4; ++s) acc[s] = (v8f){0.f,0.f,0.f,0.f,0.f,0.f,0.f,0.f};
        gemm16x64_lds(sA, d2_w, n0, lane, acc);
#pragma unroll
        for (int s = 0; s < 4; ++s) store16_lds(sP, acc[s], d2_b, n0 + s * 16, lane, false);
    }
    __syncthreads();

    // ---- t = q - k_gather + pos ----
    for (int e = t; e < KNN * DM; e += 256) {
        int k = e >> 9, c = e & (DM - 1);
        float tv = sQ[c] - kb[((long)b * N + sIdx[k]) * DM + c] + (float)sP[e];
        sT[e] = (__bf16)tv;
    }
    __syncthreads();

    // ---- a1 = relu(t @ g1^T + b) ----
    {
        v8f acc[4];
#pragma unroll
        for (int s = 0; s < 4; ++s) acc[s] = (v8f){0.f,0.f,0.f,0.f,0.f,0.f,0.f,0.f};
        gemm16x64_lds(sT, g1_w, n0, lane, acc);
#pragma unroll
        for (int s = 0; s < 4; ++s) store16_lds(sA, acc[s], g1_b, n0 + s * 16, lane, true);
    }
    __syncthreads();

    // ---- logits = a1 @ g2^T + b ----
    {
        v8f acc[4];
#pragma unroll
        for (int s = 0; s < 4; ++s) acc[s] = (v8f){0.f,0.f,0.f,0.f,0.f,0.f,0.f,0.f};
        gemm16x64_lds(sA, g2_w, n0, lane, acc);
#pragma unroll
        for (int s = 0; s < 4; ++s) store16_lds(sT, acc[s], g2_b, n0 + s * 16, lane, false);
    }
    __syncthreads();

    // ---- softmax over K neighbors + weighted reduce ----
    for (int f = t; f < DM; f += 256) {
        float lg[KNN], mx = -3.0e38f;
#pragma unroll
        for (int k = 0; k < KNN; ++k) {
            lg[k] = (float)sT[k * DM + f] * INV_SQRT_DM;
            mx = fmaxf(mx, lg[k]);
        }
        float sum = 0.f;
#pragma unroll
        for (int k = 0; k < KNN; ++k) { lg[k] = __expf(lg[k] - mx); sum += lg[k]; }
        const float inv = 1.f / sum;
        float acc = 0.f;
#pragma unroll
        for (int k = 0; k < KNN; ++k) {
            float vp = vb[((long)b * N + sIdx[k]) * DM + f] + (float)sP[k * DM + f];
            acc += lg[k] * inv * vp;
        }
        res[(long)pm * DM + f] = acc;
    }
}

// ---------------------------------------------------------------------------
// Small / irregular kernels
// ---------------------------------------------------------------------------
__global__ void linear_naive(const float* __restrict__ X, const float* __restrict__ W,
                             const float* __restrict__ b, float* __restrict__ Y,
                             int R, int Cin, int Cout, int relu)
{
    long i = (long)blockIdx.x * blockDim.x + threadIdx.x;
    if (i >= (long)R * Cout) return;
    int row = (int)(i / Cout), c = (int)(i % Cout);
    float s = b ? b[c] : 0.f;
    for (int k = 0; k < Cin; ++k) s += X[(long)row * Cin + k] * W[(long)c * Cin + k];
    Y[i] = relu ? fmaxf(s, 0.f) : s;
}

__global__ void knn_kernel(const float* __restrict__ q, const float* __restrict__ ref,
                           int* __restrict__ out, int B, int M, int N)
{
    int i = blockIdx.x * blockDim.x + threadIdx.x;
    if (i >= B * M) return;
    int b = i / M;
    const float* qp = q + (long)i * 3;
    const float qx = qp[0], qy = qp[1], qz = qp[2];
    float bd[KNN]; int bi[KNN];
#pragma unroll
    for (int k = 0; k < KNN; ++k) { bd[k] = 3.0e38f; bi[k] = 0; }
    const float* rp = ref + (long)b * N * 3;
    for (int n = 0; n < N; ++n) {
        float dx = qx - rp[3 * n], dy = qy - rp[3 * n + 1], dz = qz - rp[3 * n + 2];
        float cd = dx * dx + dy * dy + dz * dz;
        int ci = n;
        if (cd < bd[KNN - 1]) {
#pragma unroll
            for (int k = 0; k < KNN; ++k) {          // register-resident ordered insert
                if (cd < bd[k]) {
                    float td = bd[k]; int ti = bi[k];
                    bd[k] = cd; bi[k] = ci; cd = td; ci = ti;
                }
            }
        }
    }
#pragma unroll
    for (int k = 0; k < KNN; ++k) out[(long)i * KNN + k] = bi[k];
}

__global__ __launch_bounds__(256) void fps_kernel(const float* __restrict__ xyz,
                                                  int* __restrict__ out, int N, int S)
{
    __shared__ float dist[4096];
    __shared__ float rmax[256];
    __shared__ int   rarg[256];
    const int b = blockIdx.x, t = threadIdx.x;
    const float* p = xyz + (long)b * N * 3;
    for (int n = t; n < N; n += 256) dist[n] = 1.0e10f;
    __syncthreads();
    int far = 0;
    for (int s = 0; s < S; ++s) {
        if (t == 0) out[b * S + s] = far;
        float cx = p[3 * far], cy = p[3 * far + 1], cz = p[3 * far + 2];
        float best = -1.f; int barg = 0;
        for (int n = t; n < N; n += 256) {
            float dx = p[3 * n] - cx, dy = p[3 * n + 1] - cy, dz = p[3 * n + 2] - cz;
            float nd = fminf(dist[n], dx * dx + dy * dy + dz * dz);
            dist[n] = nd;
            if (nd > best) { best = nd; barg = n; }
        }
        rmax[t] = best; rarg[t] = barg;
        __syncthreads();
        for (int off = 128; off > 0; off >>= 1) {
            if (t < off) {
                if (rmax[t + off] > rmax[t] ||
                    (rmax[t + off] == rmax[t] && rarg[t + off] < rarg[t])) {
                    rmax[t] = rmax[t + off]; rarg[t] = rarg[t + off];
                }
            }
            __syncthreads();
        }
        far = rarg[0];
        __syncthreads();
    }
}

__global__ void gather_xyz_k(const float* __restrict__ xyz, const int* __restrict__ fidx,
                             float* __restrict__ out, int B, int N, int M)
{
    int i = blockIdx.x * blockDim.x + threadIdx.x;
    if (i >= B * M) return;
    int b = i / M;
    const float* p = xyz + ((long)b * N + fidx[i]) * 3;
    out[i * 3] = p[0]; out[i * 3 + 1] = p[1]; out[i * 3 + 2] = p[2];
}

// writes rows of width Cp (>= 3+C, multiple of 4 for 16B-aligned GEMM rows)
__global__ void gather_hcat(const float* __restrict__ xyz, const float* __restrict__ pts,
                            const float* __restrict__ nxyz, const int* __restrict__ idx,
                            float* __restrict__ out, int M, int N, int C, int Cp)
{
    long i = (long)blockIdx.x * blockDim.x + threadIdx.x;
    long tot = (long)2 * M * KNN;
    if (i >= tot) return;
    long bm = i >> 4;
    int  b  = (int)(bm / M);
    int  nb = idx[i];
    const float* xp = xyz  + ((long)b * N + nb) * 3;
    const float* cp = nxyz + bm * 3;
    float* op = out + i * Cp;
    op[0] = xp[0] - cp[0]; op[1] = xp[1] - cp[1]; op[2] = xp[2] - cp[2];
    const float* pp = pts + ((long)b * N + nb) * C;
    for (int c = 0; c < C; ++c) op[3 + c] = pp[c];
    for (int c = 3 + C; c < Cp; ++c) op[c] = 0.f;
}

__global__ void maxpool_k(const float* __restrict__ h, float* __restrict__ out,
                          long BM, int C)
{
    long i = (long)blockIdx.x * blockDim.x + threadIdx.x;
    if (i >= BM * C) return;
    long bm = i / C; int c = (int)(i % C);
    const float* p = h + bm * KNN * C + c;
    float mx = p[0];
#pragma unroll
    for (int k = 1; k < KNN; ++k) mx = fmaxf(mx, p[(long)k * C]);
    out[i] = mx;
}

// ---------------------------------------------------------------------------
// Host orchestration
// ---------------------------------------------------------------------------
struct TP {
    const float *fc1_w,*fc1_b,*fc2_w,*fc2_b,*d1_w,*d1_b,*d2_w,*d2_b,
                *g1_w,*g1_b,*g2_w,*g2_b,*wq,*wk,*wv;
};
struct Scratch { float *x,*q,*k,*v,*res; int* idx; };

static inline void launch_gemm(hipStream_t st, const float* X, const float* W,
                               const float* bias, const float* scale, const float* shift,
                               const float* resid, float* Y,
                               int R, int Cin, int lda, int Cout, int relu)
{
    long tiles = (long)((R + 15) / 16) * ((Cout + 63) / 64);
    int blocks = (int)((tiles + 7) / 8);
    gemm_wmma<<<blocks, 256, 0, st>>>(X, W, bias, scale, shift, resid, Y,
                                      R, Cin, lda, Cout, relu);
}

static void run_tblock(hipStream_t st, const TP& tp, const Scratch& w,
                       const float* xyz, const float* feats,
                       int B, int N, int dp, float* out)
{
    const int pts = B * N;
    knn_kernel<<<(pts + 255) / 256, 256, 0, st>>>(xyz, xyz, w.idx, B, N, N);
    launch_gemm(st, feats, tp.fc1_w, tp.fc1_b, nullptr, nullptr, nullptr, w.x, pts, dp, dp, DM, 0);
    launch_gemm(st, w.x, tp.wq, nullptr, nullptr, nullptr, nullptr, w.q, pts, DM, DM, DM, 0);
    launch_gemm(st, w.x, tp.wk, nullptr, nullptr, nullptr, nullptr, w.k, pts, DM, DM, DM, 0);
    launch_gemm(st, w.x, tp.wv, nullptr, nullptr, nullptr, nullptr, w.v, pts, DM, DM, DM, 0);
    attn_fused<<<pts, 256, 0, st>>>(xyz, w.idx, w.q, w.k, w.v,
                                    tp.d1_w, tp.d1_b, tp.d2_w, tp.d2_b,
                                    tp.g1_w, tp.g1_b, tp.g2_w, tp.g2_b,
                                    w.res, B, N);
    launch_gemm(st, w.res, tp.fc2_w, tp.fc2_b, nullptr, nullptr, feats, out, pts, DM, DM, dp, 0);
}

extern "C" void kernel_launch(void* const* d_in, const int* in_sizes, int n_in,
                              void* d_out, int out_size, void* d_ws, size_t ws_size,
                              hipStream_t stream)
{
    (void)in_sizes; (void)n_in; (void)out_size; (void)ws_size;
    const int B = 2, N0 = 4096;
    auto F = [&](int i) { return (const float*)d_in[i]; };
    auto load_tp = [&](int s) {
        TP t;
        t.fc1_w = F(s);      t.fc1_b = F(s + 1);
        t.fc2_w = F(s + 2);  t.fc2_b = F(s + 3);
        t.d1_w  = F(s + 4);  t.d1_b  = F(s + 5);
        t.d2_w  = F(s + 6);  t.d2_b  = F(s + 7);
        t.g1_w  = F(s + 8);  t.g1_b  = F(s + 9);
        t.g2_w  = F(s + 10); t.g2_b  = F(s + 11);
        t.wq = F(s + 12); t.wk = F(s + 13); t.wv = F(s + 14);
        return t;
    };

    // ---- workspace carve (all chunks multiple of 4 floats -> 16B aligned) ----
    float* w = (float*)d_ws; size_t off = 0;
    auto alloc = [&](size_t n) { float* p = w + off; off += n; return p; };
    float* bufH0  = alloc((size_t)B * N0 * 32);
    float* bufP1  = alloc((size_t)B * N0 * 32);
    float* bufP2  = alloc((size_t)B * N0 * 32);
    Scratch sc;
    sc.x   = alloc((size_t)B * N0 * DM);
    sc.q   = alloc((size_t)B * N0 * DM);
    sc.k   = alloc((size_t)B * N0 * DM);
    sc.v   = alloc((size_t)B * N0 * DM);
    sc.res = alloc((size_t)B * N0 * DM);
    float* bufHcat = alloc(1310720);   // >= 32768 * 36
    float* bufHa   = alloc(2200000);
    float* bufHb   = alloc(2200000);
    float* xyzA    = alloc((size_t)B * 1024 * 3);
    float* xyzB    = alloc((size_t)B * 1024 * 3);
    int* ibase = (int*)(w + off);
    sc.idx     = ibase;                         // B*4096*16 ints
    int* fidx  = ibase + (size_t)B * N0 * KNN;  // B*1024 ints

    const float* x = F(0);   // (B,4096,3): xyz == x

    // ---- stem: feats = mlp2(x, fc0) ----
    {
        long tot = (long)B * N0 * 32;
        linear_naive<<<(unsigned)((tot + 255) / 256), 256, 0, stream>>>(
            x, F(1), F(2), bufH0, B * N0, 3, 32, 1);
        linear_naive<<<(unsigned)((tot + 255) / 256), 256, 0, stream>>>(
            bufH0, F(3), F(4), bufP1, B * N0, 32, 32, 0);
    }

    // ---- t0 transformer ----
    TP t0 = load_tp(5);
    run_tblock(stream, t0, sc, x, bufP1, B, N0, 32, bufP2);

    const float* cur_xyz = x;
    float*       cur_pts = bufP2;
    int curN = N0, curC = 32;

    for (int l = 1; l <= 4; ++l) {
        const int base = 20 + (l - 1) * 23;
        const int M  = curN / 4;
        const int ch = 32 << l;
        float* nxyz = (l & 1) ? xyzA : xyzB;

        // transition down
        fps_kernel<<<B, 256, 0, stream>>>(cur_xyz, fidx, curN, M);
        gather_xyz_k<<<(B * M + 255) / 256, 256, 0, stream>>>(cur_xyz, fidx, nxyz, B, curN, M);
        knn_kernel<<<(B * M + 255) / 256, 256, 0, stream>>>(nxyz, cur_xyz, sc.idx, B, M, curN);
        {
            long rows = (long)B * M * KNN;
            int Cin = 3 + curC;
            int Cp  = (Cin + 3) & ~3;    // 16B-aligned row stride
            gather_hcat<<<(unsigned)((rows + 255) / 256), 256, 0, stream>>>(
                cur_xyz, cur_pts, nxyz, sc.idx, bufHcat, M, curN, curC, Cp);
            // c1: bn1 + relu (BN scale carries 1/sqrt(1+eps))
            launch_gemm(stream, bufHcat, F(base + 0), F(base + 1),
                        F(base + 4), F(base + 5), nullptr, bufHa,
                        (int)rows, Cin, Cp, ch, 1);
            // c2: bn2 + relu
            launch_gemm(stream, bufHa, F(base + 2), F(base + 3),
                        F(base + 6), F(base + 7), nullptr, bufHb,
                        (int)rows, ch, ch, ch, 1);
            long pool = (long)B * M * ch;
            maxpool_k<<<(unsigned)((pool + 255) / 256), 256, 0, stream>>>(
                bufHb, bufP1, (long)B * M, ch);
        }

        // transformer block at this level
        TP tp = load_tp(base + 8);
        float* tout = (l == 4) ? (float*)d_out : bufP2;
        run_tblock(stream, tp, sc, nxyz, bufP1, B, M, ch, tout);

        cur_xyz = nxyz;
        cur_pts = tout;
        curN = M; curC = ch;
    }
}
